// kmeans_52828097741284
// MI455X (gfx1250) — matrix operand order repaired and checked by reading, not compiled, not run on hardware
//
#include <hip/hip_runtime.h>

#define D 64
#define K 256
#define ITERS 10
#define TILE_PTS 128
#define TILES_PER_BLOCK 16
#define BLOCK 256

typedef __attribute__((ext_vector_type(16))) __bf16 v16bf;
typedef __attribute__((ext_vector_type(8)))  float  v8f;

__device__ __forceinline__ unsigned short f32_to_bf16_bits(float f) {
    unsigned u = __builtin_bit_cast(unsigned, f);
    unsigned r = u + 0x7FFFu + ((u >> 16) & 1u);   // round-to-nearest-even
    return (unsigned short)(r >> 16);
}

union BfFrag {
    v16bf v;
    unsigned short s[16];
    uint4 q[2];
};

// centroids0 = x[init_idx]
__global__ __launch_bounds__(BLOCK)
void kmeans_init_kernel(const float* __restrict__ x, const int* __restrict__ idx,
                        float* __restrict__ c) {
    int gid = blockIdx.x * BLOCK + threadIdx.x;    // < K*D
    int k = gid >> 6, d = gid & (D - 1);
    c[gid] = x[idx[k] * D + d];
}

// per-iteration: bf16 centroids, ||c||^2, zero sums/counts
__global__ __launch_bounds__(BLOCK)
void kmeans_prep_kernel(const float* __restrict__ c,
                        unsigned short* __restrict__ cbf,
                        float* __restrict__ cnorm,
                        float* __restrict__ sums,
                        float* __restrict__ counts) {
    __shared__ float sh[BLOCK];
    const int tid = threadIdx.x;
    const int gid = blockIdx.x * BLOCK + tid;      // grid = 64 blocks -> K*D elems
    float cv = c[gid];
    cbf[gid]  = f32_to_bf16_bits(cv);
    sums[gid] = 0.0f;
    if (blockIdx.x == 0) counts[tid] = 0.0f;       // K == 256 == BLOCK
    sh[tid] = cv * cv;
    __syncthreads();
    if (tid < 4) {                                 // 4 clusters per block
        float s = 0.0f;
        #pragma unroll
        for (int j = 0; j < D; ++j) s += sh[tid * D + j];
        cnorm[blockIdx.x * 4 + tid] = s;
    }
}

// fused: distances (WMMA bf16) -> argmin -> LDS partial segment-sum -> global flush
__global__ __launch_bounds__(BLOCK)
void kmeans_assign_kernel(const float* __restrict__ x,
                          const unsigned short* __restrict__ cbf,
                          const float* __restrict__ cnorm,
                          float* __restrict__ sums_g,
                          float* __restrict__ counts_g,
                          int N) {
    __shared__ unsigned short cent_s[K * D];   // 32 KB  (B matrices, bf16 row-major)
    __shared__ float cnorm_s[K];               //  1 KB
    __shared__ float xtile[TILE_PTS * D];      // 32 KB  (exact f32 tile)
    __shared__ float sums_s[K * D];            // 64 KB  (per-block partial sums)
    __shared__ float counts_s[K];              //  1 KB
    __shared__ int   assign_s[TILE_PTS];       // 0.5 KB

    const int tid = threadIdx.x;

    for (int j = tid; j < K * D; j += BLOCK) sums_s[j] = 0.0f;
    counts_s[tid] = 0.0f;

    {   // stage centroids (bf16) into LDS: 2048 uint4
        uint4* dst = (uint4*)cent_s;
        const uint4* src = (const uint4*)cbf;
        #pragma unroll
        for (int i = 0; i < (K * D * 2 / 16) / BLOCK; ++i)
            dst[tid + i * BLOCK] = src[tid + i * BLOCK];
    }
    cnorm_s[tid] = cnorm[tid];
    __syncthreads();

    const int lane  = tid & 31;
    const int wave  = tid >> 5;       // 0..7 -> 16-row strip
    const int l15   = lane & 15;
    const int lhalf = lane >> 4;

    // ||c||^2 for this lane's column in every column tile: invariant across point tiles
    float cnr[K / 16];
    #pragma unroll
    for (int t = 0; t < K / 16; ++t) cnr[t] = cnorm_s[t * 16 + l15];

    for (int tt = 0; tt < TILES_PER_BLOCK; ++tt) {
        const int tileStart = (blockIdx.x * TILES_PER_BLOCK + tt) * TILE_PTS;
        if (tileStart >= N) break;                       // uniform across block
        const int plim = min(TILE_PTS, N - tileStart);

        {   // stage x tile (f32), zero-pad past N
            const float4* xg = (const float4*)x;
            float4* xd = (float4*)xtile;
            const int base4 = tileStart * (D / 4);
            #pragma unroll
            for (int i = 0; i < (TILE_PTS * D / 4) / BLOCK; ++i) {
                int f4 = tid + i * BLOCK;                // 0..2047, 16 per point
                int p = f4 >> 4;
                float4 v = make_float4(0.f, 0.f, 0.f, 0.f);
                if (p < plim) v = xg[base4 + f4];
                xd[f4] = v;
            }
        }
        __syncthreads();

        // A fragments (two K-steps of 32) for this wave's strip.
        // 16-bit A layout: lane(M)=l15; lanes<16 hold K {0..7,16..23}, lanes>=16 {8..15,24..31}
        const int row = wave * 16 + l15;
        BfFrag a0, a1;
        {
            const float* ap0 = &xtile[row * D + 0 * 32 + lhalf * 8];
            const float* ap1 = &xtile[row * D + 1 * 32 + lhalf * 8];
            #pragma unroll
            for (int j = 0; j < 8; ++j) {
                a0.s[j]     = f32_to_bf16_bits(ap0[j]);
                a0.s[8 + j] = f32_to_bf16_bits(ap0[16 + j]);
                a1.s[j]     = f32_to_bf16_bits(ap1[j]);
                a1.s[8 + j] = f32_to_bf16_bits(ap1[16 + j]);
            }
        }

        float mv[8];
        int   mi[8];
        #pragma unroll
        for (int i = 0; i < 8; ++i) { mv[i] = 3.4e38f; mi[i] = 0; }

        for (int t = 0; t < K / 16; ++t) {
            // B layout: lane(N)=l15; lanes<16 K {0..7,16..23} of the step, lanes>=16 K {8..15,24..31}
            const int n = t * 16 + l15;
            BfFrag b0, b1;
            const uint4* bp0 = (const uint4*)&cent_s[n * D + 0 * 32 + lhalf * 8];
            const uint4* bp1 = (const uint4*)&cent_s[n * D + 1 * 32 + lhalf * 8];
            b0.q[0] = bp0[0]; b0.q[1] = bp0[2];
            b1.q[0] = bp1[0]; b1.q[1] = bp1[2];

            v8f acc = {};
            acc = __builtin_amdgcn_wmma_f32_16x16x32_bf16(false, a0.v, false, b0.v,
                                                          (short)0, acc, false, false);
            acc = __builtin_amdgcn_wmma_f32_16x16x32_bf16(false, a1.v, false, b1.v,
                                                          (short)0, acc, false, false);
            const float cn = cnr[t];
            #pragma unroll
            for (int i = 0; i < 8; ++i) {
                float dd = cn - 2.0f * acc[i];           // ||c||^2 - 2 x.c (x_sq is row-const)
                bool take = dd < mv[i];
                mv[i] = take ? dd : mv[i];
                mi[i] = take ? n : mi[i];
            }
        }

        // Branchless argmin across the 16 lanes sharing each row half (C layout:
        // rows 0-7 in lanes 0-15, rows 8-15 in lanes 16-31).
        // Tie-break: lowest index (jnp.argmin semantics). Non-short-circuit ops
        // keep this as v_cmp + v_cndmask with no exec-mask branches.
        #pragma unroll
        for (int off = 1; off < 16; off <<= 1) {
            #pragma unroll
            for (int i = 0; i < 8; ++i) {
                float ov = __shfl_xor(mv[i], off, 32);
                int   oi = __shfl_xor(mi[i], off, 32);
                int take = (int)(ov < mv[i]) | ((int)(ov == mv[i]) & (int)(oi < mi[i]));
                mv[i] = take ? ov : mv[i];
                mi[i] = take ? oi : mi[i];
            }
        }
        if (l15 == 0) {
            #pragma unroll
            for (int i = 0; i < 8; ++i)
                assign_s[wave * 16 + lhalf * 8 + i] = mi[i];
        }
        __syncthreads();

        {   // per-block segment-sum: 2 threads per point, 32 dims each, LDS atomics
            const int p = tid >> 1;
            const int dbase = (tid & 1) * (D / 2);
            if (p < plim) {
                const int a = assign_s[p];
                const float* xp = &xtile[p * D + dbase];
                float* sp = &sums_s[a * D + dbase];
                #pragma unroll
                for (int j = 0; j < D / 2; ++j)
                    atomicAdd(&sp[j], xp[j]);
                if (dbase == 0) atomicAdd(&counts_s[a], 1.0f);
            }
        }
        __syncthreads();
    }

    // flush partials (skip zeros to cut global atomic traffic)
    for (int j = tid; j < K * D; j += BLOCK) {
        float v = sums_s[j];
        if (v != 0.0f) atomicAdd(&sums_g[j], v);
    }
    {
        float c = counts_s[tid];
        if (c != 0.0f) atomicAdd(&counts_g[tid], c);
    }
}

// new_c = counts>0 ? sums/counts : old_c
__global__ __launch_bounds__(BLOCK)
void kmeans_update_kernel(const float* __restrict__ sums,
                          const float* __restrict__ counts,
                          const float* __restrict__ cin,
                          float* __restrict__ cout) {
    int gid = blockIdx.x * BLOCK + threadIdx.x;
    int k = gid >> 6;
    float cnt = counts[k];
    cout[gid] = (cnt > 0.0f) ? (sums[gid] / cnt) : cin[gid];
}

extern "C" void kernel_launch(void* const* d_in, const int* in_sizes, int n_in,
                              void* d_out, int out_size, void* d_ws, size_t ws_size,
                              hipStream_t stream) {
    const float* x        = (const float*)d_in[0];
    const int*   init_idx = (const int*)d_in[1];
    const int N = in_sizes[0] / D;

    char* w = (char*)d_ws;
    float* c_ws   = (float*)w; w += (size_t)K * D * sizeof(float);
    float* sums   = (float*)w; w += (size_t)K * D * sizeof(float);
    float* counts = (float*)w; w += (size_t)K * sizeof(float);
    float* cnorm  = (float*)w; w += (size_t)K * sizeof(float);
    unsigned short* cbf = (unsigned short*)w;

    float* dout = (float*)d_out;

    dim3 blk(BLOCK);
    dim3 grid_kd((K * D) / BLOCK);   // 64 blocks

    kmeans_init_kernel<<<grid_kd, blk, 0, stream>>>(x, init_idx, c_ws);

    const int ptsPerBlock = TILE_PTS * TILES_PER_BLOCK;
    const int nblkAssign = (N + ptsPerBlock - 1) / ptsPerBlock;   // 245 for N=500000
    for (int it = 0; it < ITERS; ++it) {
        kmeans_prep_kernel<<<grid_kd, blk, 0, stream>>>(c_ws, cbf, cnorm, sums, counts);
        kmeans_assign_kernel<<<dim3(nblkAssign), blk, 0, stream>>>(x, cbf, cnorm, sums, counts, N);
        kmeans_update_kernel<<<grid_kd, blk, 0, stream>>>(sums, counts, c_ws,
                                                          (it == ITERS - 1) ? dout : c_ws);
    }
}
